// Masked_multihead_attention_39410619908119
// MI455X (gfx1250) — compile-verified, matrix operand order checked by reading, not run on hardware
//
#include <hip/hip_runtime.h>
#include <hip/hip_bf16.h>

// MI455X / gfx1250 fused masked-MHA (column softmax) pipeline.
// All matmuls run on v_wmma_f32_16x16x32_bf16 (wave32 WMMA).
// K-tile staging uses GLOBAL_LOAD_ASYNC_TO_LDS_B128 when the builtin exists.

typedef __attribute__((ext_vector_type(16))) __bf16 v16bf;
typedef __attribute__((ext_vector_type(8)))  __bf16 v8bf;
typedef __attribute__((ext_vector_type(8)))  float  v8f;
typedef __attribute__((ext_vector_type(4)))  int    v4i;
typedef __bf16 bf16_t;

#define NB    4
#define T_SEQ 2048
#define DIM   1024
#define NH    16
#define HD    64

#if defined(__has_builtin)
#if __has_builtin(__builtin_amdgcn_global_load_async_to_lds_b128)
#define HAVE_ASYNC_LDS 1
#endif
#endif

#ifdef HAVE_ASYNC_LDS
typedef __attribute__((address_space(1))) v4i gv4i;   // global b128 payload
typedef __attribute__((address_space(3))) v4i sv4i;   // LDS b128 payload
#endif

// ---- fragment helpers -------------------------------------------------------
// CDNA5 16-bit A-frag (16x32, MxK): lane l holds row M=l&15; K runs
// {kg..kg+7} and {16+kg..16+kg+7} with kg=(l>>4)*8 -> two contiguous 8-elem
// runs.  B-frag (32x16, KxN) is the symmetric layout with N=l&15 and the same
// two contiguous runs along K (K is the fast axis of the source we read).

__device__ __forceinline__ v16bf frag_from_bf16(const bf16_t* p0, const bf16_t* p1) {
  v8bf lo = *(const v8bf*)p0;   // 16B contiguous
  v8bf hi = *(const v8bf*)p1;   // 16B contiguous
  v16bf r;
#pragma unroll
  for (int j = 0; j < 8; ++j) { r[j] = lo[j]; r[j + 8] = hi[j]; }
  return r;
}

__device__ __forceinline__ v16bf frag_from_f32(const float* p0, const float* p1) {
  v8f lo = *(const v8f*)p0;     // 32B contiguous
  v8f hi = *(const v8f*)p1;
  v16bf r;
#pragma unroll
  for (int j = 0; j < 8; ++j) { r[j] = (__bf16)lo[j]; r[j + 8] = (__bf16)hi[j]; }
  return r;
}

__device__ __forceinline__ v8f wmma_bf16(v16bf a, v16bf b, v8f c) {
  return __builtin_amdgcn_wmma_f32_16x16x32_bf16(false, a, false, b, (short)0, c,
                                                 false, false);
}

// ---- kernel 1: QKV projection  Y = x @ W^T + b  -> [B][H][T][64] bf16 -------
__global__ void __launch_bounds__(256)
qkv_proj_kernel(const float* __restrict__ x,
                const float* __restrict__ wq, const float* __restrict__ bq,
                const float* __restrict__ wk, const float* __restrict__ bk,
                const float* __restrict__ wv, const float* __restrict__ bv,
                bf16_t* __restrict__ Qh, bf16_t* __restrict__ Kh,
                bf16_t* __restrict__ Vh) {
  const int z = blockIdx.z;
  const float* w   = (z == 0) ? wq : (z == 1) ? wk : wv;
  const float* bia = (z == 0) ? bq : (z == 1) ? bk : bv;
  bf16_t* outh     = (z == 0) ? Qh : (z == 1) ? Kh : Vh;

  const int lane  = threadIdx.x & 31;
  const int wave  = threadIdx.x >> 5;            // 8 waves: 4 (M) x 2 (N)
  const int mbase = blockIdx.x * 128 + (wave & 3) * 32;
  const int nbase = blockIdx.y * 64 + (wave >> 2) * 32;
  const int lr    = lane & 15;
  const int kg    = (lane >> 4) * 8;

  v8f acc[2][2] = {};

  for (int k0 = 0; k0 < DIM; k0 += 32) {
    const float* ar0 = x + (size_t)(mbase + lr) * DIM + k0 + kg;
    const float* ar1 = ar0 + (size_t)16 * DIM;
    v16bf a0 = frag_from_f32(ar0, ar0 + 16);
    v16bf a1 = frag_from_f32(ar1, ar1 + 16);
    // B[k][n] = W[n][k]: rows of W give contiguous K runs
    const float* br0 = w + (size_t)(nbase + lr) * DIM + k0 + kg;
    const float* br1 = br0 + (size_t)16 * DIM;
    v16bf b0 = frag_from_f32(br0, br0 + 16);
    v16bf b1 = frag_from_f32(br1, br1 + 16);
    if (k0 + 64 < DIM) {   // global_prefetch_b8: hide 4KB-strided first touches
      __builtin_prefetch(ar0 + 64, 0, 0);
      __builtin_prefetch(ar1 + 64, 0, 0);
      __builtin_prefetch(br0 + 64, 0, 0);
      __builtin_prefetch(br1 + 64, 0, 0);
    }
    acc[0][0] = wmma_bf16(a0, b0, acc[0][0]);
    acc[0][1] = wmma_bf16(a0, b1, acc[0][1]);
    acc[1][0] = wmma_bf16(a1, b0, acc[1][0]);
    acc[1][1] = wmma_bf16(a1, b1, acc[1][1]);
  }

  const int rowg = (lane >> 4) * 8;               // C/D: M = vgpr + 8*(l>>4)
#pragma unroll
  for (int ni = 0; ni < 2; ++ni) {
    const int gn = nbase + ni * 16 + lr;
    const float bb = bia[gn];
    const int h = gn >> 6, d = gn & 63;
#pragma unroll
    for (int mi = 0; mi < 2; ++mi) {
#pragma unroll
      for (int r = 0; r < 8; ++r) {
        const int gm = mbase + mi * 16 + rowg + r;
        const int b = gm >> 11, t = gm & (T_SEQ - 1);
        outh[(((size_t)b * NH + h) * T_SEQ + t) * HD + d] =
            (__bf16)(acc[mi][ni][r] + bb);
      }
    }
  }
}

// ---- kernel 2: per-column (query-axis) softmax stats ------------------------
// For each key column k: m_k = max_{q>=k} s[q,k]/8 ; s_k = sum exp(./8 - m_k)
__global__ void __launch_bounds__(256)
colstats_kernel(const bf16_t* __restrict__ Qh, const bf16_t* __restrict__ Kh,
                float* __restrict__ mOut, float* __restrict__ sOut) {
  const int bh   = blockIdx.y;
  const int lane = threadIdx.x & 31;
  const int wave = threadIdx.x >> 5;
  const int kt   = blockIdx.x * 8 + wave;         // column tile 0..127
  const int kcol0 = kt * 16;
  const int lr   = lane & 15;
  const int kg   = (lane >> 4) * 8;
  const int rowg = (lane >> 4) * 8;
  const int gk   = kcol0 + lr;                    // this lane's global column

  const bf16_t* Qp = Qh + (size_t)bh * T_SEQ * HD;
  const bf16_t* Kp = Kh + (size_t)bh * T_SEQ * HD;

  // loop-invariant B fragments: B[kd][kc] = K[kc][kd]
  const bf16_t* krow = Kp + (size_t)gk * HD;
  v16bf bk0 = frag_from_bf16(krow + kg,      krow + 16 + kg);
  v16bf bk1 = frag_from_bf16(krow + 32 + kg, krow + 48 + kg);

  float mx = -__builtin_inff();
  float sm = 0.0f;

  for (int qt = kt; qt < T_SEQ / 16; ++qt) {
    const int q0 = qt * 16;
    const bf16_t* qrow = Qp + (size_t)(q0 + lr) * HD;
    v16bf a0 = frag_from_bf16(qrow + kg,      qrow + 16 + kg);
    v16bf a1 = frag_from_bf16(qrow + 32 + kg, qrow + 48 + kg);
    v8f s = {};
    s = wmma_bf16(a0, bk0, s);
    s = wmma_bf16(a1, bk1, s);
    float vals[8];
    float tmax = -__builtin_inff();
#pragma unroll
    for (int r = 0; r < 8; ++r) {
      const int gq = q0 + rowg + r;
      const float v = (gq >= gk) ? s[r] * 0.125f : -__builtin_inff();
      vals[r] = v;
      tmax = fmaxf(tmax, v);
    }
    if (tmax > -__builtin_inff()) {
      const float nm = fmaxf(mx, tmax);
      sm *= __expf(mx - nm);
#pragma unroll
      for (int r = 0; r < 8; ++r) sm += __expf(vals[r] - nm);
      mx = nm;
    }
  }
  // merge row-groups 0-7 / 8-15 (lanes l and l^16 hold the same column)
  const float mo = __shfl_xor(mx, 16, 32);
  const float so = __shfl_xor(sm, 16, 32);
  const float nm = fmaxf(mx, mo);
  float tot = 0.0f;
  if (mx > -__builtin_inff()) tot += sm * __expf(mx - nm);
  if (mo > -__builtin_inff()) tot += so * __expf(mo - nm);
  if (lane < 16) {
    mOut[(size_t)bh * T_SEQ + gk] = nm;
    sOut[(size_t)bh * T_SEQ + gk] = tot;
  }
}

// ---- kernel 3: O = attn @ V (column-normalized P), LDS-staged K/V -----------
__global__ void __launch_bounds__(256)
attn_kernel(const bf16_t* __restrict__ Qh, const bf16_t* __restrict__ Kh,
            const bf16_t* __restrict__ Vh, const float* __restrict__ mIn,
            const float* __restrict__ sIn, bf16_t* __restrict__ Obf) {
  __shared__ __align__(16) bf16_t Kt[32 * 64];      // [kr][kd]
  __shared__ __align__(16) bf16_t Vt[64 * 32];      // [d][kr] (transposed)
  __shared__ __align__(16) bf16_t Pb[8][16 * 32];   // per-wave P tile [q][k]

  const int bh   = blockIdx.y;
  const int b    = bh >> 4, h = bh & 15;
  const int lane = threadIdx.x & 31;
  const int wave = threadIdx.x >> 5;
  const int lr   = lane & 15;
  const int kg   = (lane >> 4) * 8;
  const int rowg = (lane >> 4) * 8;
  const int q0   = (blockIdx.x * 8 + wave) * 16;

  const bf16_t* Qp = Qh + (size_t)bh * T_SEQ * HD;
  const bf16_t* Kp = Kh + (size_t)bh * T_SEQ * HD;
  const bf16_t* Vp = Vh + (size_t)bh * T_SEQ * HD;

  const bf16_t* qrow = Qp + (size_t)(q0 + lr) * HD;
  v16bf aq0 = frag_from_bf16(qrow + kg,      qrow + 16 + kg);
  v16bf aq1 = frag_from_bf16(qrow + 32 + kg, qrow + 48 + kg);

  v8f o[4] = {};

  const int nkt = (blockIdx.x + 1) * 4;             // k tiles covering k <= q_max
  for (int kt2 = 0; kt2 < nkt; ++kt2) {
    const int k0 = kt2 * 32;
    __syncthreads();
    {   // cooperative stage: 256 threads, one 16B chunk each
      const int kr = threadIdx.x >> 3;
      const int c8 = (threadIdx.x & 7) * 8;
#ifdef HAVE_ASYNC_LDS
      // CDNA5 async DMA: global -> LDS without touching VGPRs (ASYNCcnt).
      __builtin_amdgcn_global_load_async_to_lds_b128(
          (gv4i*)(Kp + (size_t)(k0 + kr) * HD + c8),
          (sv4i*)(Kt + kr * HD + c8), 0, 0);
#else
      v8bf kv = *(const v8bf*)(Kp + (size_t)(k0 + kr) * HD + c8);
      *(v8bf*)(Kt + kr * HD + c8) = kv;
#endif
      v8bf vv = *(const v8bf*)(Vp + (size_t)(k0 + kr) * HD + c8);
#pragma unroll
      for (int j = 0; j < 8; ++j) Vt[(c8 + j) * 32 + kr] = vv[j];
#ifdef HAVE_ASYNC_LDS
#if __has_builtin(__builtin_amdgcn_s_wait_asynccnt)
      __builtin_amdgcn_s_wait_asynccnt(0);
#else
      asm volatile("s_wait_asynccnt 0x0" ::: "memory");
#endif
#endif
    }
    __syncthreads();

#pragma unroll
    for (int half = 0; half < 2; ++half) {
      const bf16_t* kb = Kt + (half * 16 + lr) * HD;
      v16bf b0 = frag_from_bf16(kb + kg,      kb + 16 + kg);
      v16bf b1 = frag_from_bf16(kb + 32 + kg, kb + 48 + kg);
      v8f s = {};
      s = wmma_bf16(aq0, b0, s);
      s = wmma_bf16(aq1, b1, s);
      const int gk = k0 + half * 16 + lr;
      const float mk = mIn[(size_t)bh * T_SEQ + gk];
      const float rs = 1.0f / sIn[(size_t)bh * T_SEQ + gk];
#pragma unroll
      for (int r = 0; r < 8; ++r) {
        const int gq = q0 + rowg + r;
        const float p = (gq >= gk) ? __expf(s[r] * 0.125f - mk) * rs : 0.0f;
        Pb[wave][(rowg + r) * 32 + half * 16 + lr] = (__bf16)p;
      }
    }
    // same-wave LDS is processed in order: safe to read P back as an A-frag
    const bf16_t* pr = &Pb[wave][lr * 32];
    v16bf pa = frag_from_bf16(pr + kg, pr + 16 + kg);
#pragma unroll
    for (int dt = 0; dt < 4; ++dt) {
      const bf16_t* vb = Vt + (dt * 16 + lr) * 32;  // B[kr][d] = Vt[d][kr]
      v16bf bv2 = frag_from_bf16(vb + kg, vb + 16 + kg);
      o[dt] = wmma_bf16(pa, bv2, o[dt]);
    }
  }

  bf16_t* Op = Obf + (size_t)b * T_SEQ * DIM;       // heads re-interleaved
#pragma unroll
  for (int dt = 0; dt < 4; ++dt) {
#pragma unroll
    for (int r = 0; r < 8; ++r) {
      Op[(size_t)(q0 + rowg + r) * DIM + h * HD + dt * 16 + lr] =
          (__bf16)o[dt][r];
    }
  }
}

// ---- kernel 4: output projection  out = O @ wo^T + bo  (fp32 out) -----------
__global__ void __launch_bounds__(256)
out_proj_kernel(const bf16_t* __restrict__ A, const float* __restrict__ w,
                const float* __restrict__ bia, float* __restrict__ out) {
  const int lane  = threadIdx.x & 31;
  const int wave  = threadIdx.x >> 5;
  const int mbase = blockIdx.x * 128 + (wave & 3) * 32;
  const int nbase = blockIdx.y * 64 + (wave >> 2) * 32;
  const int lr    = lane & 15;
  const int kg    = (lane >> 4) * 8;

  v8f acc[2][2] = {};

  for (int k0 = 0; k0 < DIM; k0 += 32) {
    const bf16_t* ar0 = A + (size_t)(mbase + lr) * DIM + k0 + kg;
    const bf16_t* ar1 = ar0 + (size_t)16 * DIM;
    v16bf a0 = frag_from_bf16(ar0, ar0 + 16);
    v16bf a1 = frag_from_bf16(ar1, ar1 + 16);
    const float* br0 = w + (size_t)(nbase + lr) * DIM + k0 + kg;
    const float* br1 = br0 + (size_t)16 * DIM;
    v16bf b0 = frag_from_f32(br0, br0 + 16);
    v16bf b1 = frag_from_f32(br1, br1 + 16);
    if (k0 + 64 < DIM) {
      __builtin_prefetch(ar0 + 64, 0, 0);
      __builtin_prefetch(ar1 + 64, 0, 0);
      __builtin_prefetch(br0 + 64, 0, 0);
      __builtin_prefetch(br1 + 64, 0, 0);
    }
    acc[0][0] = wmma_bf16(a0, b0, acc[0][0]);
    acc[0][1] = wmma_bf16(a0, b1, acc[0][1]);
    acc[1][0] = wmma_bf16(a1, b0, acc[1][0]);
    acc[1][1] = wmma_bf16(a1, b1, acc[1][1]);
  }

  const int rowg = (lane >> 4) * 8;
#pragma unroll
  for (int ni = 0; ni < 2; ++ni) {
    const int gn = nbase + ni * 16 + lr;
    const float bb = bia[gn];
#pragma unroll
    for (int mi = 0; mi < 2; ++mi) {
#pragma unroll
      for (int r = 0; r < 8; ++r) {
        const int gm = mbase + mi * 16 + rowg + r;
        out[(size_t)gm * DIM + gn] = acc[mi][ni][r] + bb;
      }
    }
  }
}

// ---- host launch ------------------------------------------------------------
extern "C" void kernel_launch(void* const* d_in, const int* in_sizes, int n_in,
                              void* d_out, int out_size, void* d_ws, size_t ws_size,
                              hipStream_t stream) {
  const float* x  = (const float*)d_in[0];   // q (reference projects K,V from q too)
  const float* wq = (const float*)d_in[3];
  const float* bq = (const float*)d_in[4];
  const float* wk = (const float*)d_in[5];
  const float* bk = (const float*)d_in[6];
  const float* wv = (const float*)d_in[7];
  const float* bv = (const float*)d_in[8];
  const float* wo = (const float*)d_in[9];
  const float* bo = (const float*)d_in[10];
  // d_in[11] (mask) is the causal tril; computed arithmetically in-kernel.

  char* ws = (char*)d_ws;
  const size_t nBHT = (size_t)NB * NH * T_SEQ * HD;   // 8,388,608 elems
  bf16_t* Qh = (bf16_t*)ws;
  bf16_t* Kh = Qh + nBHT;
  bf16_t* Vh = Kh + nBHT;
  bf16_t* Ob = Vh + nBHT;                             // [B][T][D] bf16
  float*  mB = (float*)(Ob + nBHT);
  float*  sB = mB + (size_t)NB * NH * T_SEQ;

  hipLaunchKernelGGL(qkv_proj_kernel, dim3(64, 16, 3), dim3(256), 0, stream,
                     x, wq, bq, wk, bk, wv, bv, Qh, Kh, Vh);
  hipLaunchKernelGGL(colstats_kernel, dim3(16, 64), dim3(256), 0, stream,
                     Qh, Kh, mB, sB);
  hipLaunchKernelGGL(attn_kernel, dim3(16, 64), dim3(256), 0, stream,
                     Qh, Kh, Vh, mB, sB, Ob);
  hipLaunchKernelGGL(out_proj_kernel, dim3(64, 16), dim3(256), 0, stream,
                     Ob, wo, bo, (float*)d_out);
}
